// SelfAttention_17179869184702
// MI455X (gfx1250) — matrix operand order repaired and checked by reading, not compiled, not run on hardware
//
#include <hip/hip_runtime.h>
#include <hip/hip_fp16.h>

// ---------------------------------------------------------------------------
// Self-attention for MI455X (gfx1250), wave32 + WMMA f16->f32 + TDM staging.
//   B=2, T=2048, D=2048, H=16, HD=128.
// Pipeline: cast -> 3x GEMM (QKV, TDM-staged weights) -> RoPE + V-transpose
//           -> flash attention (WMMA) -> output GEMM.
// ---------------------------------------------------------------------------

typedef __attribute__((ext_vector_type(16))) _Float16 v16h;
typedef __attribute__((ext_vector_type(8)))  _Float16 v8h;
typedef __attribute__((ext_vector_type(8)))  float    v8f;
typedef __attribute__((ext_vector_type(4)))  unsigned int u32x4;
typedef __attribute__((ext_vector_type(8)))  int      i32x8;
typedef __attribute__((ext_vector_type(4)))  int      i32x4;

#define BB 2
#define TT 2048
#define DD 2048
#define HH 16
#define HD 128
#define NN (BB * TT)   // 4096 rows

__device__ __forceinline__ v16h cat8(v8h lo, v8h hi) {
  return __builtin_shufflevector(lo, hi, 0,1,2,3,4,5,6,7,8,9,10,11,12,13,14,15);
}

__device__ __forceinline__ v8f wmma_f16(v16h a, v16h b, v8f c) {
  return __builtin_amdgcn_wmma_f32_16x16x32_f16(false, a, false, b, (short)0, c,
                                                false, false);
}

// ---------------------------------------------------------------------------
// TDM: DMA a 64-row x 32-col f16 tile of W (row stride K=2048 elements) into
// LDS at lds_off, padding 16B after every 64B row so LDS row stride is 80B.
// D# per CDNA5 ISA ch.8: group0 {count,lds,global,type=2}; group1 {data_size,
// pad, tensor dims, tile dims, stride}; groups 2/3 zero (2-D tile).
// This toolchain exposes the 6-arg builtin: (g0, g1, g2, g3, g_extra, cpol).
// ---------------------------------------------------------------------------
__device__ __forceinline__ void tdm_load_wtile(const _Float16* gptr,
                                               unsigned int lds_off) {
  const unsigned long long ga = (unsigned long long)(uintptr_t)gptr;
  u32x4 g0;
  g0.x = 1u;                                   // count=1 (valid user D#)
  g0.y = lds_off;                              // [63:32]  lds_addr
  g0.z = (unsigned int)ga;                     // [95:64]  global_addr lo
  g0.w = (unsigned int)((ga >> 32) & 0x01FFFFFFu) | 0x80000000u; // +type=2
  i32x8 g1;
  g1[0] = (int)((1u << 16)      // data_size = 2 bytes
              | (1u << 20)      // pad_enable
              | (3u << 22)      // pad_interval: every 16 DWORDs (64B)
              | (3u << 25));    // pad_amount: 4 DWORDs (16B)
  g1[1] = (int)(2048u << 16);   // [63:48] tensor_dim0[15:0] = 2048
  g1[2] = (int)(2048u << 16);   // [79:64] dim0 hi=0; [95:80] tensor_dim1 lo
  g1[3] = (int)(32u << 16);     // [111:96] dim1 hi=0; [127:112] tile_dim0=32
  g1[4] = 64;                   // [143:128] tile_dim1=64; tile_dim2=0
  g1[5] = 2048;                 // tensor_dim0_stride = K (elements)
  g1[6] = 0;
  g1[7] = 0;
  const i32x4 z4 = {0, 0, 0, 0};
  const i32x8 z8 = {0, 0, 0, 0, 0, 0, 0, 0};
  __builtin_amdgcn_tensor_load_to_lds(g0, g1, z4, z4, z8, 0);
}

// ---------------------------------------------------------------------------
// Kernel 1: f32 -> f16 cast, 4 elements/thread
// ---------------------------------------------------------------------------
__global__ void cast_f32_f16(const float* __restrict__ s,
                             _Float16* __restrict__ d, int n) {
  int i = (blockIdx.x * blockDim.x + threadIdx.x) * 4;
  if (i + 3 < n) {
    float4 v = *(const float4*)(s + i);
    d[i + 0] = (_Float16)v.x;
    d[i + 1] = (_Float16)v.y;
    d[i + 2] = (_Float16)v.z;
    d[i + 3] = (_Float16)v.w;
  }
}

// ---------------------------------------------------------------------------
// Kernel 2: C = A(NxK) * W(MxK)^T.  Block = 8 waves = 128 rows x 64 cols.
// W tiles (shared by all 8 waves) are TDM-staged into LDS, double-buffered:
// wave 0 issues the DMA for chunk kc+1, waits tensorcnt<=1 (chunk kc done),
// block barrier publishes the tile.  Grid is exact; no early returns.
// ---------------------------------------------------------------------------
__global__ __launch_bounds__(256)
void gemm_nt_wmma(const _Float16* __restrict__ A,
                  const _Float16* __restrict__ W,
                  _Float16* __restrict__ out16,   // may be null
                  float*    __restrict__ out32,   // may be null
                  int N, int M, int K) {
  __shared__ _Float16 wtile[2][64 * 40];   // 80B row stride (64B data+16B pad)
  const int lane = threadIdx.x & 31;
  const int wid  = threadIdx.x >> 5;
  const int ln   = lane & 15;
  const int hl   = lane >> 4;
  const int rowBase = (blockIdx.x * 8 + wid) * 16;
  const int colBase = blockIdx.y * 64;

  const unsigned int lds0 = (unsigned int)(uintptr_t)&wtile[0][0];
  const unsigned int lds1 = (unsigned int)(uintptr_t)&wtile[1][0];
  const _Float16* wbase = W + (size_t)colBase * K;
  const int NK = K / 32;

  if (wid == 0) tdm_load_wtile(wbase, lds0);   // prologue: chunk 0 -> buf0

  v8f acc[4] = {};
  for (int kc = 0; kc < NK; ++kc) {
    const int cur = kc & 1;
    if (wid == 0) {
      // issue next chunk into the other buffer (dummy reload on last iter)
      const int knext = (kc + 1 < NK) ? (kc + 1) * 32 : 0;
      tdm_load_wtile(wbase + knext, cur ? lds0 : lds1);
      __builtin_amdgcn_s_wait_tensorcnt(1);    // chunk kc resident
    }
    __syncthreads();

    const int k0 = kc * 32;
    // A fragment: lane row = ln; K halves split by hl (ISA 16-bit A layout)
    const _Float16* ap = A + (size_t)(rowBase + ln) * K + k0 + hl * 8;
    v16h afrag = cat8(*(const v8h*)ap, *(const v8h*)(ap + 16));
    const _Float16* wt = wtile[cur];
#pragma unroll
    for (int t = 0; t < 4; ++t) {
      // B fragment: lane column = ln; K halves split by hl (matches A)
      const _Float16* lp = wt + (t * 16 + ln) * 40 + hl * 8;
      v16h bfrag = cat8(*(const v8h*)lp, *(const v8h*)(lp + 16));
      acc[t] = wmma_f16(afrag, bfrag, acc[t]);
    }
    __syncthreads();   // protect buf[cur] before it is re-targeted next iter
  }

  // C layout: element r -> (row = r + 8*hl, col = ln)
  if (out16) {
#pragma unroll
    for (int t = 0; t < 4; ++t)
#pragma unroll
      for (int r = 0; r < 8; ++r)
        out16[(size_t)(rowBase + r + 8 * hl) * M + colBase + t * 16 + ln] =
            (_Float16)acc[t][r];
  }
  if (out32) {
#pragma unroll
    for (int t = 0; t < 4; ++t)
#pragma unroll
      for (int r = 0; r < 8; ++r)
        out32[(size_t)(rowBase + r + 8 * hl) * M + colBase + t * 16 + ln] =
            acc[t][r];
  }
}

// ---------------------------------------------------------------------------
// Kernel 3: RoPE on q,k (layout [N,D], per-head slices contiguous) and
// transpose v into vT[b][h][hd][t] so PV B-fragments are contiguous.
// ---------------------------------------------------------------------------
__global__ void rope_vt_kernel(const _Float16* __restrict__ qf,
                               const _Float16* __restrict__ kf,
                               const _Float16* __restrict__ vf,
                               const float* __restrict__ cosT,
                               const float* __restrict__ sinT,
                               _Float16* __restrict__ qr,
                               _Float16* __restrict__ kr,
                               _Float16* __restrict__ vT) {
  const int n = blockIdx.x;        // 0..NN-1
  const int b = n / TT, t = n % TT;
  for (int d = threadIdx.x; d < DD; d += blockDim.x) {
    const int h = d / HD, hd = d % HD;
    const float c = cosT[t * HD + hd];
    const float s = sinT[t * HD + hd];
    const size_t base = (size_t)n * DD;
    const int pd = (hd < 64) ? d + 64 : d - 64;
    const float rs = (hd < 64) ? -1.0f : 1.0f;
    float q  = (float)qf[base + d],  k  = (float)kf[base + d];
    float qp = (float)qf[base + pd], kp = (float)kf[base + pd];
    qr[base + d] = (_Float16)(c * q + rs * qp * s);
    kr[base + d] = (_Float16)(c * k + rs * kp * s);
    vT[(((size_t)b * HH + h) * HD + hd) * TT + t] = vf[base + d];
  }
}

// ---------------------------------------------------------------------------
// Kernel 4: causal flash attention.  One wave owns a 16-row query block of
// one (b,h); loops over 32-wide key blocks.  8 WMMA for S, 8 WMMA for PV per
// key block.  P goes C-layout -> A-layout via a private LDS tile guarded by
// s_wait_dscnt (waves have different trip counts, so no block barrier).
// ---------------------------------------------------------------------------
__global__ __launch_bounds__(256)
void flash_attn_wmma(const _Float16* __restrict__ q,   // [N,D] roped
                     const _Float16* __restrict__ k,   // [N,D] roped
                     const _Float16* __restrict__ vT,  // [B,H,HD,T]
                     _Float16* __restrict__ ctx) {     // [N,D]
  __shared__ _Float16 ldsP[8][16][40];   // per-wave 16x32 tile, padded rows
  const int lane = threadIdx.x & 31;
  const int wid  = threadIdx.x >> 5;
  const int ln   = lane & 15;
  const int hl   = lane >> 4;

  const int w   = blockIdx.x * 8 + wid;          // global wave id
  const int b   = w / (HH * (TT / 16));
  const int rem = w % (HH * (TT / 16));
  const int h   = rem / (TT / 16);
  const int qb  = rem % (TT / 16);
  const size_t rowq = (size_t)b * TT + (size_t)qb * 16;

  // Q fragments for the whole head dim (4 x K=32)
  v16h qfrag[4];
#pragma unroll
  for (int c = 0; c < 4; ++c) {
    const _Float16* p = q + (rowq + ln) * DD + h * HD + c * 32 + hl * 8;
    qfrag[c] = cat8(*(const v8h*)p, *(const v8h*)(p + 16));
  }

  const float scale = 0.08838834764831845f;  // 1/sqrt(128)
  float m[8], l[8];
  v8f acc[8] = {};
#pragma unroll
  for (int r = 0; r < 8; ++r) { m[r] = -1e30f; l[r] = 0.0f; }

  const int nkb = ((qb * 16 + 15) >> 5) + 1;   // causal: key blocks of 32
  for (int kb = 0; kb < nkb; ++kb) {
    // ---- S = Q K^T for two 16-col tiles
    v8f S[2] = {};
#pragma unroll
    for (int tile = 0; tile < 2; ++tile)
#pragma unroll
      for (int c = 0; c < 4; ++c) {
        const _Float16* kp = k +
            ((size_t)b * TT + kb * 32 + tile * 16 + ln) * DD +
            h * HD + c * 32 + hl * 8;
        S[tile] = wmma_f16(qfrag[c], cat8(*(const v8h*)kp, *(const v8h*)(kp + 16)), S[tile]);
      }
    // ---- scale + causal mask + online softmax update
    float mnew[8];
#pragma unroll
    for (int r = 0; r < 8; ++r) {
      const int qg = qb * 16 + r + 8 * hl;
      float s0 = S[0][r] * scale;
      float s1 = S[1][r] * scale;
      if (kb * 32 + ln      > qg) s0 = -1e30f;
      if (kb * 32 + 16 + ln > qg) s1 = -1e30f;
      S[0][r] = s0; S[1][r] = s1;
      float rm = fmaxf(s0, s1);
      rm = fmaxf(rm, __shfl_xor(rm, 1));
      rm = fmaxf(rm, __shfl_xor(rm, 2));
      rm = fmaxf(rm, __shfl_xor(rm, 4));
      rm = fmaxf(rm, __shfl_xor(rm, 8));
      mnew[r] = fmaxf(m[r], rm);
    }
#pragma unroll
    for (int r = 0; r < 8; ++r) {
      const float p0 = __expf(S[0][r] - mnew[r]);
      const float p1 = __expf(S[1][r] - mnew[r]);
      const float alpha = __expf(m[r] - mnew[r]);
      float rs = p0 + p1;
      rs += __shfl_xor(rs, 1);
      rs += __shfl_xor(rs, 2);
      rs += __shfl_xor(rs, 4);
      rs += __shfl_xor(rs, 8);
      l[r] = l[r] * alpha + rs;
      m[r] = mnew[r];
#pragma unroll
      for (int d8 = 0; d8 < 8; ++d8) acc[d8][r] = acc[d8][r] * alpha;
      // scatter P (C layout) into the wave-private LDS tile
      ldsP[wid][r + 8 * hl][ln]      = (_Float16)p0;
      ldsP[wid][r + 8 * hl][16 + ln] = (_Float16)p1;
    }
    // same-wave LDS ordering: DS ops complete in order; just drain DScnt
    asm volatile("s_wait_dscnt 0x0" ::: "memory");
    __builtin_amdgcn_wave_barrier();
    // gather P in A-fragment layout
    v16h pfrag = cat8(*(const v8h*)&ldsP[wid][ln][hl * 8],
                      *(const v8h*)&ldsP[wid][ln][16 + hl * 8]);
    // ---- acc += P V  (8 output tiles across HD=128)
#pragma unroll
    for (int d8 = 0; d8 < 8; ++d8) {
      const _Float16* vp = vT +
          (((size_t)b * HH + h) * HD + d8 * 16 + ln) * TT +
          kb * 32 + hl * 8;
      acc[d8] = wmma_f16(pfrag, cat8(*(const v8h*)vp, *(const v8h*)(vp + 16)), acc[d8]);
    }
  }
  // ---- normalize and store ctx in [N, D] f16 for the output projection
#pragma unroll
  for (int r = 0; r < 8; ++r) {
    const float inv = 1.0f / l[r];
    const size_t row = rowq + r + 8 * hl;
#pragma unroll
    for (int d8 = 0; d8 < 8; ++d8)
      ctx[row * DD + h * HD + d8 * 16 + ln] = (_Float16)(acc[d8][r] * inv);
  }
}

// ---------------------------------------------------------------------------
// Host launcher
// ---------------------------------------------------------------------------
extern "C" void kernel_launch(void* const* d_in, const int* in_sizes, int n_in,
                              void* d_out, int out_size, void* d_ws,
                              size_t ws_size, hipStream_t stream) {
  (void)in_sizes; (void)n_in; (void)out_size; (void)ws_size;
  const float* x        = (const float*)d_in[0];
  const float* Wq       = (const float*)d_in[1];
  const float* Wk       = (const float*)d_in[2];
  const float* Wv       = (const float*)d_in[3];
  const float* Wo       = (const float*)d_in[4];
  const float* rope_cos = (const float*)d_in[5];
  const float* rope_sin = (const float*)d_in[6];
  float* out = (float*)d_out;

  // workspace carve-up (f16 buffers), ~160 MB total
  char* p = (char*)d_ws;
  const size_t szND = (size_t)NN * DD * sizeof(_Float16);  // 16 MB
  const size_t szDD = (size_t)DD * DD * sizeof(_Float16);  //  8 MB
  _Float16* xb   = (_Float16*)p; p += szND;
  _Float16* wq16 = (_Float16*)p; p += szDD;
  _Float16* wk16 = (_Float16*)p; p += szDD;
  _Float16* wv16 = (_Float16*)p; p += szDD;
  _Float16* wo16 = (_Float16*)p; p += szDD;
  _Float16* qf   = (_Float16*)p; p += szND;
  _Float16* kf   = (_Float16*)p; p += szND;
  _Float16* vf   = (_Float16*)p; p += szND;
  _Float16* qr   = (_Float16*)p; p += szND;
  _Float16* kr   = (_Float16*)p; p += szND;
  _Float16* vTb  = (_Float16*)p; p += szND;
  _Float16* ctx  = (_Float16*)p; p += szND;

  // 1) casts
  cast_f32_f16<<<(NN * DD) / 1024, 256, 0, stream>>>(x, xb, NN * DD);
  cast_f32_f16<<<(DD * DD) / 1024, 256, 0, stream>>>(Wq, wq16, DD * DD);
  cast_f32_f16<<<(DD * DD) / 1024, 256, 0, stream>>>(Wk, wk16, DD * DD);
  cast_f32_f16<<<(DD * DD) / 1024, 256, 0, stream>>>(Wv, wv16, DD * DD);
  cast_f32_f16<<<(DD * DD) / 1024, 256, 0, stream>>>(Wo, wo16, DD * DD);

  // 2) QKV projections (WMMA + TDM weight staging)
  dim3 gg(NN / 128, DD / 64);
  gemm_nt_wmma<<<gg, 256, 0, stream>>>(xb, wq16, qf, nullptr, NN, DD, DD);
  gemm_nt_wmma<<<gg, 256, 0, stream>>>(xb, wk16, kf, nullptr, NN, DD, DD);
  gemm_nt_wmma<<<gg, 256, 0, stream>>>(xb, wv16, vf, nullptr, NN, DD, DD);

  // 3) RoPE + V transpose
  rope_vt_kernel<<<NN, 256, 0, stream>>>(qf, kf, vf, rope_cos, rope_sin,
                                         qr, kr, vTb);

  // 4) flash attention (WMMA): B*H*(T/16) = 4096 waves / 8 per block
  flash_attn_wmma<<<(BB * HH * (TT / 16)) / 8, 256, 0, stream>>>(qr, kr, vTb,
                                                                 ctx);

  // 5) output projection -> f32 result
  gemm_nt_wmma<<<gg, 256, 0, stream>>>(ctx, wo16, nullptr, out, NN, DD, DD);
}
